// RelativePositionalAttention_90529320665340
// MI455X (gfx1250) — matrix operand order repaired and checked by reading, not compiled, not run on hardware
//
#include <hip/hip_runtime.h>

// ---------------------------------------------------------------------------
// RelativePositionalAttention for MI455X (gfx1250, wave32, WMMA 16x16x32 f16)
//   BATCH=2, SEQ=2048, D_MODEL=1024, H=16, DK=64, MAX_REL=32
// ---------------------------------------------------------------------------

typedef __attribute__((ext_vector_type(16))) _Float16 v16h;
typedef __attribute__((ext_vector_type(8)))  float    v8f;
typedef int v4i __attribute__((vector_size(16)));   // matches async-LDS builtin

#define BATCH   2
#define SEQ     2048
#define DMODEL  1024
#define NHEADS  16
#define DK      64
#define MAXREL  32
#define NROWS   (BATCH * SEQ)          // 4096
#define BH      (BATCH * NHEADS)       // 32
#define QE_LD   72                     // padded 65 -> 72 floats per row
#define LDSST   40                     // LDS row stride in halves (pad vs 32)
#define KTST    72                     // K-tile LDS row stride in halves (pad vs 64)

#if __has_builtin(__builtin_amdgcn_global_load_async_to_lds_b128) && \
    __has_builtin(__builtin_amdgcn_s_wait_asynccnt)
#define HAVE_ASYNC_LDS 1
#else
#define HAVE_ASYNC_LDS 0
#endif

static __device__ inline v8f v8f_zero() {
  v8f z;
#pragma unroll
  for (int i = 0; i < 8; ++i) z[i] = 0.0f;
  return z;
}

#if HAVE_ASYNC_LDS
// Generic->LDS(AS3): low 32 bits of a generic LDS pointer are the DS offset
// (ISA §10.2 aperture rules). Generic->global(AS1) is value-identity.
// Builtin wants int4-vector pointees (per hipcc diagnostic), non-const.
static __device__ inline __attribute__((address_space(1))) v4i*
to_gbl4(const _Float16* p) {
  return (__attribute__((address_space(1))) v4i*)(uintptr_t)p;
}
static __device__ inline __attribute__((address_space(3))) v4i*
to_lds4(_Float16* p) {
  return (__attribute__((address_space(3))) v4i*)(unsigned)(uintptr_t)p;
}
#endif

// A-fragment (16x32, f16) per-lane K index:  lanes 0-15 row M=lane, lanes 16-31 same M.
//   k = hi*16 + i (contiguous)
// B-fragment (32x16, f16): lane holds column N=lane&15;  k = hi*16 + i (contiguous).
// C/D f32:   element r -> row M = r + hi*8, col N = lane&15.

// ---------------------------------------------------------------------------
// Projection: Y(f16, head-split [BH, SEQ, DK]) = scale * (X[f32] @ W[f32]^T)
// grid (NROWS/128, DMODEL/128), block 256 (8 waves: 4 along n x 2 along m).
// ---------------------------------------------------------------------------
__global__ __launch_bounds__(256) void proj_qkv_kernel(
    const float* __restrict__ X, const float* __restrict__ W,
    _Float16* __restrict__ Y, float scale) {
  __shared__ __align__(16) _Float16 Ah[128 * LDSST];
  __shared__ __align__(16) _Float16 Bh[128 * LDSST];

  const int lane = threadIdx.x & 31;
  const int wave = threadIdx.x >> 5;
  const int nq   = lane & 15;
  const int hi   = lane >> 4;

  const int n0 = blockIdx.x * 128;
  const int m0 = blockIdx.y * 128;
  const int wn = (wave & 3) * 32;   // 2 n-subtiles of 16
  const int wm = (wave >> 2) * 64;  // 4 m-subtiles of 16

  v8f acc[2][4];
#pragma unroll
  for (int a = 0; a < 2; ++a)
#pragma unroll
    for (int b = 0; b < 4; ++b) acc[a][b] = v8f_zero();

  const int ldr = threadIdx.x >> 1;        // 0..127
  const int ldc = (threadIdx.x & 1) * 16;  // 0 or 16

  for (int kk = 0; kk < DMODEL; kk += 32) {
    {
      const float* xp = X + (size_t)(n0 + ldr) * DMODEL + kk + ldc;
      const float* wp = W + (size_t)(m0 + ldr) * DMODEL + kk + ldc;
#pragma unroll
      for (int i = 0; i < 16; ++i) Ah[ldr * LDSST + ldc + i] = (_Float16)xp[i];
#pragma unroll
      for (int i = 0; i < 16; ++i) Bh[ldr * LDSST + ldc + i] = (_Float16)wp[i];
    }
    __syncthreads();

    v16h af[2], bf[4];
#pragma unroll
    for (int a = 0; a < 2; ++a) {
      const _Float16* p = Ah + (wn + a * 16 + nq) * LDSST;
#pragma unroll
      for (int i = 0; i < 16; ++i)
        af[a][i] = p[(hi << 4) + i];
    }
#pragma unroll
    for (int b = 0; b < 4; ++b) {
      const _Float16* p = Bh + (wm + b * 16 + nq) * LDSST + hi * 16;
#pragma unroll
      for (int i = 0; i < 16; ++i) bf[b][i] = p[i];
    }
#pragma unroll
    for (int a = 0; a < 2; ++a)
#pragma unroll
      for (int b = 0; b < 4; ++b)
        acc[a][b] = __builtin_amdgcn_wmma_f32_16x16x32_f16(
            false, af[a], false, bf[b], (short)0, acc[a][b], false, false);
    __syncthreads();
  }

  // store: head-split f16 layout [bh][s][dk]
#pragma unroll
  for (int a = 0; a < 2; ++a)
#pragma unroll
    for (int b = 0; b < 4; ++b)
#pragma unroll
      for (int r = 0; r < 8; ++r) {
        int n  = n0 + wn + a * 16 + r + hi * 8;   // global row (b*SEQ + s)
        int m  = m0 + wm + b * 16 + nq;           // output column
        int bb = n >> 11;                         // / SEQ
        int s  = n & (SEQ - 1);
        int h  = m >> 6;                          // / DK
        int dk = m & (DK - 1);
        Y[(((size_t)(bb * NHEADS + h)) * SEQ + s) * DK + dk] =
            (_Float16)(acc[a][b][r] * scale);
      }
}

// ---------------------------------------------------------------------------
// Output projection: out(f32) = X(f16)[NROWS x DMODEL] @ W^T + bias
// ---------------------------------------------------------------------------
__global__ __launch_bounds__(256) void proj_out_kernel(
    const _Float16* __restrict__ X, const float* __restrict__ W,
    const float* __restrict__ bias, float* __restrict__ out) {
  __shared__ __align__(16) _Float16 Ah[128 * LDSST];
  __shared__ __align__(16) _Float16 Bh[128 * LDSST];

  const int lane = threadIdx.x & 31;
  const int wave = threadIdx.x >> 5;
  const int nq   = lane & 15;
  const int hi   = lane >> 4;

  const int n0 = blockIdx.x * 128;
  const int m0 = blockIdx.y * 128;
  const int wn = (wave & 3) * 32;
  const int wm = (wave >> 2) * 64;

  v8f acc[2][4];
#pragma unroll
  for (int a = 0; a < 2; ++a)
#pragma unroll
    for (int b = 0; b < 4; ++b) acc[a][b] = v8f_zero();

  const int ldr = threadIdx.x >> 1;
  const int ldc = (threadIdx.x & 1) * 16;

  for (int kk = 0; kk < DMODEL; kk += 32) {
    {
      const _Float16* xp = X + (size_t)(n0 + ldr) * DMODEL + kk + ldc;
      const float*    wp = W + (size_t)(m0 + ldr) * DMODEL + kk + ldc;
#pragma unroll
      for (int i = 0; i < 16; ++i) Ah[ldr * LDSST + ldc + i] = xp[i];
#pragma unroll
      for (int i = 0; i < 16; ++i) Bh[ldr * LDSST + ldc + i] = (_Float16)wp[i];
    }
    __syncthreads();

    v16h af[2], bf[4];
#pragma unroll
    for (int a = 0; a < 2; ++a) {
      const _Float16* p = Ah + (wn + a * 16 + nq) * LDSST;
#pragma unroll
      for (int i = 0; i < 16; ++i)
        af[a][i] = p[(hi << 4) + i];
    }
#pragma unroll
    for (int b = 0; b < 4; ++b) {
      const _Float16* p = Bh + (wm + b * 16 + nq) * LDSST + hi * 16;
#pragma unroll
      for (int i = 0; i < 16; ++i) bf[b][i] = p[i];
    }
#pragma unroll
    for (int a = 0; a < 2; ++a)
#pragma unroll
      for (int b = 0; b < 4; ++b)
        acc[a][b] = __builtin_amdgcn_wmma_f32_16x16x32_f16(
            false, af[a], false, bf[b], (short)0, acc[a][b], false, false);
    __syncthreads();
  }

#pragma unroll
  for (int a = 0; a < 2; ++a)
#pragma unroll
    for (int b = 0; b < 4; ++b)
#pragma unroll
      for (int r = 0; r < 8; ++r) {
        int n = n0 + wn + a * 16 + r + hi * 8;
        int m = m0 + wm + b * 16 + nq;
        out[(size_t)n * DMODEL + m] = acc[a][b][r] + bias[m];
      }
}

// ---------------------------------------------------------------------------
// QE[row][r] = Qh[row] . rel_emb[r]   (Q already carries 1/sqrt(dk))
// rows = BH*SEQ, r in [0,65), stored padded to QE_LD.
// ---------------------------------------------------------------------------
__global__ void qe_kernel(const _Float16* __restrict__ Qh,
                          const float* __restrict__ rel,
                          float* __restrict__ QE) {
  size_t tid = (size_t)blockIdx.x * blockDim.x + threadIdx.x;
  int    r   = (int)(tid % 80);
  size_t row = tid / 80;
  if (r >= 2 * MAXREL + 1 || row >= (size_t)BH * SEQ) return;
  const _Float16* q = Qh + row * DK;
  const float*    e = rel + (size_t)r * DK;
  float acc = 0.0f;
#pragma unroll
  for (int k = 0; k < DK; ++k) acc += (float)q[k] * e[k];
  QE[row * QE_LD + r] = acc;
}

// ---------------------------------------------------------------------------
// Flash attention with relative-position bias.
// grid = BH * (SEQ/64) blocks, 128 threads (4 waves, each owns 16 query rows).
// K tile staged via GLOBAL_LOAD_ASYNC_TO_LDS_B128 (ASYNCcnt) when available.
// ---------------------------------------------------------------------------
__global__ __launch_bounds__(128) void attn_kernel(
    const _Float16* __restrict__ Qh, const _Float16* __restrict__ Kh,
    const _Float16* __restrict__ Vh, const float* __restrict__ QE,
    _Float16* __restrict__ Oh) {
  __shared__ __align__(16) _Float16 VT[DK][LDSST];      // transposed V tile (32 keys)
  __shared__ __align__(16) _Float16 Kt[32][KTST];       // K tile (row-major, 32 keys)
  __shared__ __align__(16) _Float16 Pt[4][16][LDSST];   // per-wave P tile 16x32

  const int lane = threadIdx.x & 31;
  const int wave = threadIdx.x >> 5;
  const int nq   = lane & 15;
  const int hi   = lane >> 4;

  const int tiles = SEQ / 64;                       // 32
  const int bh    = blockIdx.x / tiles;             // 0..31
  const int q0    = (blockIdx.x % tiles) * 64 + wave * 16;

  const _Float16* Qb  = Qh + (size_t)bh * SEQ * DK;
  const _Float16* Kb  = Kh + (size_t)bh * SEQ * DK;
  const _Float16* Vb  = Vh + (size_t)bh * SEQ * DK;
  const float*    QEb = QE + (size_t)bh * SEQ * QE_LD;

  // Q fragments for this wave's 16 rows (k = 0..63 -> two 16x32 A-frags)
  v16h qf[2];
#pragma unroll
  for (int ks = 0; ks < 2; ++ks) {
    const _Float16* p = Qb + (size_t)(q0 + nq) * DK + ks * 32;
#pragma unroll
    for (int i = 0; i < 16; ++i)
      qf[ks][i] = p[(hi << 4) + i];
  }

  float m_r[8], l_r[8];
  v8f   oacc[4];
#pragma unroll
  for (int r = 0; r < 8; ++r) { m_r[r] = -1e30f; l_r[r] = 0.0f; }
#pragma unroll
  for (int d = 0; d < 4; ++d) oacc[d] = v8f_zero();

  for (int jb = 0; jb < SEQ; jb += 32) {
    __syncthreads();  // previous iteration's LDS reads complete

    // ---- stage K tile (32 rows x 64 halves) into LDS ----
#if HAVE_ASYNC_LDS
    {
      // 256 16-byte chunks; 2 async copies per thread, no VGPR round-trip.
#pragma unroll
      for (int u = 0; u < 2; ++u) {
        int id = threadIdx.x * 2 + u;        // 0..255
        int jr = id >> 3;                    // row 0..31
        int c8 = (id & 7) << 3;              // half-col 0,8,...,56
        __builtin_amdgcn_global_load_async_to_lds_b128(
            to_gbl4(Kb + (size_t)(jb + jr) * DK + c8),
            to_lds4(&Kt[jr][c8]), 0, 0);
      }
    }
#else
    {
      int jr = threadIdx.x >> 2;             // 0..31
      int c0 = (threadIdx.x & 3) << 4;       // 0,16,32,48
      const _Float16* kp = Kb + (size_t)(jb + jr) * DK + c0;
#pragma unroll
      for (int i = 0; i < 16; ++i) Kt[jr][c0 + i] = kp[i];
    }
#endif

    // ---- cooperative transposed stage of V[jb..jb+32)[0..64) ----
    {
      int jr = threadIdx.x >> 2;            // 0..31
      int c0 = (threadIdx.x & 3) << 4;      // 0,16,32,48
      const _Float16* vp = Vb + (size_t)(jb + jr) * DK + c0;
#pragma unroll
      for (int i = 0; i < 16; ++i) VT[c0 + i][jr] = vp[i];
    }
    if (jb + 32 < SEQ) {  // hint next tiles toward the caches
      __builtin_prefetch(Kb + (size_t)(jb + 32 + (threadIdx.x >> 2)) * DK, 0, 1);
      __builtin_prefetch(Vb + (size_t)(jb + 32 + (threadIdx.x >> 2)) * DK, 0, 1);
    }
#if HAVE_ASYNC_LDS
    __builtin_amdgcn_s_wait_asynccnt(0);
#endif
    __syncthreads();

    // scores for two 16-col key subtiles
    v8f sc[2];
#pragma unroll
    for (int jt = 0; jt < 2; ++jt) {
      v8f c = v8f_zero();
      const int j0 = jb + jt * 16;
#pragma unroll
      for (int ks = 0; ks < 2; ++ks) {
        v16h kf;  // B-frag: column = key row j0+nq, k contiguous (from LDS)
        const _Float16* p = &Kt[jt * 16 + nq][ks * 32 + hi * 16];
#pragma unroll
        for (int i = 0; i < 16; ++i) kf[i] = p[i];
        c = __builtin_amdgcn_wmma_f32_16x16x32_f16(
            false, qf[ks], false, kf, (short)0, c, false, false);
      }
      // relative-position bias gather (already scaled via Q)
#pragma unroll
      for (int r = 0; r < 8; ++r) {
        int irow = q0 + r + hi * 8;
        int dlt  = (j0 + nq) - irow;
        dlt = dlt < -MAXREL ? -MAXREL : (dlt > MAXREL ? MAXREL : dlt);
        c[r] += QEb[(size_t)irow * QE_LD + (dlt + MAXREL)];
      }
      sc[jt] = c;
    }

    // online softmax update, P -> LDS (f16)
    float corr_r[8];
#pragma unroll
    for (int r = 0; r < 8; ++r) {
      float mt = fmaxf(sc[0][r], sc[1][r]);
#pragma unroll
      for (int m = 1; m < 16; m <<= 1) mt = fmaxf(mt, __shfl_xor(mt, m, 32));
      float mnew = fmaxf(m_r[r], mt);
      float p0   = __expf(sc[0][r] - mnew);
      float p1   = __expf(sc[1][r] - mnew);
      float rs   = p0 + p1;
#pragma unroll
      for (int m = 1; m < 16; m <<= 1) rs += __shfl_xor(rs, m, 32);
      float corr = __expf(m_r[r] - mnew);
      l_r[r]    = l_r[r] * corr + rs;
      m_r[r]    = mnew;
      corr_r[r] = corr;
      Pt[wave][r + hi * 8][nq]      = (_Float16)p0;
      Pt[wave][r + hi * 8][16 + nq] = (_Float16)p1;
    }
#pragma unroll
    for (int d = 0; d < 4; ++d)
#pragma unroll
      for (int r = 0; r < 8; ++r) oacc[d][r] *= corr_r[r];

    __builtin_amdgcn_wave_barrier();  // keep LDS store->load order (same wave)

    // P A-fragment (16 rows x 32 keys)
    v16h pf;
    {
      const _Float16* p = &Pt[wave][nq][0];
#pragma unroll
      for (int i = 0; i < 16; ++i)
        pf[i] = p[(hi << 4) + i];
    }
    // O += P @ V  (4 d-tiles, B-frags from transposed V in LDS)
#pragma unroll
    for (int d = 0; d < 4; ++d) {
      v16h vf;
      const _Float16* p = &VT[d * 16 + nq][hi * 16];
#pragma unroll
      for (int i = 0; i < 16; ++i) vf[i] = p[i];
      oacc[d] = __builtin_amdgcn_wmma_f32_16x16x32_f16(
          false, pf, false, vf, (short)0, oacc[d], false, false);
    }
  }

  // finalize: O /= l, store f16 into [B*SEQ, H*DK] (row-major for out-proj)
  const int b = bh >> 4, h = bh & (NHEADS - 1);
#pragma unroll
  for (int d = 0; d < 4; ++d)
#pragma unroll
    for (int r = 0; r < 8; ++r) {
      int row = q0 + r + hi * 8;
      int col = h * DK + d * 16 + nq;
      Oh[(size_t)(b * SEQ + row) * DMODEL + col] =
          (_Float16)(oacc[d][r] / l_r[r]);
    }
}

// ---------------------------------------------------------------------------
extern "C" void kernel_launch(void* const* d_in, const int* in_sizes, int n_in,
                              void* d_out, int out_size, void* d_ws, size_t ws_size,
                              hipStream_t stream) {
  (void)in_sizes; (void)n_in; (void)out_size; (void)ws_size;
  const float* query = (const float*)d_in[0];
  const float* key_  = (const float*)d_in[1];
  const float* value = (const float*)d_in[2];
  const float* Wq    = (const float*)d_in[3];
  const float* Wk    = (const float*)d_in[4];
  const float* Wv    = (const float*)d_in[5];
  const float* Wo    = (const float*)d_in[6];
  const float* bo    = (const float*)d_in[7];
  const float* rel   = (const float*)d_in[8];
  float* out = (float*)d_out;

  char* ws = (char*)d_ws;
  _Float16* Qh = (_Float16*)ws; ws += (size_t)BH * SEQ * DK * sizeof(_Float16);
  _Float16* Kh = (_Float16*)ws; ws += (size_t)BH * SEQ * DK * sizeof(_Float16);
  _Float16* Vh = (_Float16*)ws; ws += (size_t)BH * SEQ * DK * sizeof(_Float16);
  float*    QE = (float*)ws;    ws += (size_t)BH * SEQ * QE_LD * sizeof(float);
  _Float16* Oh = (_Float16*)ws; ws += (size_t)NROWS * DMODEL * sizeof(_Float16);

  dim3 gg(NROWS / 128, DMODEL / 128), bb(256);
  // Q gets 1/sqrt(DK) folded in, so both content and positional scores are scaled.
  proj_qkv_kernel<<<gg, bb, 0, stream>>>(query, Wq, Qh, 0.125f);
  proj_qkv_kernel<<<gg, bb, 0, stream>>>(key_,  Wk, Kh, 1.0f);
  proj_qkv_kernel<<<gg, bb, 0, stream>>>(value, Wv, Vh, 1.0f);

  qe_kernel<<<((size_t)BH * SEQ * 80) / 256, 256, 0, stream>>>(Qh, rel, QE);

  attn_kernel<<<BH * (SEQ / 64), 128, 0, stream>>>(Qh, Kh, Vh, QE, Oh);

  proj_out_kernel<<<gg, bb, 0, stream>>>(Oh, Wo, bo, out);
}
